// SinkhornCausalAttention_48747878809991
// MI455X (gfx1250) — compile-verified, matrix operand order checked by reading, not run on hardware
//
#include <hip/hip_runtime.h>
#include <hip/hip_bf16.h>

typedef __attribute__((ext_vector_type(16))) __bf16 v16bf;
typedef __attribute__((ext_vector_type(8)))  float  v8f;

#define BH 32
#define T_LEN 8192
#define DH 64
#define BSZ 128
#define NBUCK 64
#define NEGMAX (-3.3895313892515355e38f)

__device__ __forceinline__ __bf16 f2bf(float x) {
    union { float f; unsigned u; } c; c.f = x;
    unsigned r = c.u + 0x7fffu + ((c.u >> 16) & 1u);
    unsigned short h = (unsigned short)(r >> 16);
    return __builtin_bit_cast(__bf16, h);
}

__device__ __forceinline__ v16bf ldfrag(const __bf16* p) {
    union { uint4 u[2]; v16bf f; } r;
    r.u[0] = *(const uint4*)(p);
    r.u[1] = *(const uint4*)(p + 16);
    return r.f;
}

__device__ __forceinline__ float redmax16(float v) {
    v = fmaxf(v, __shfl_xor(v, 1, 32));
    v = fmaxf(v, __shfl_xor(v, 2, 32));
    v = fmaxf(v, __shfl_xor(v, 4, 32));
    v = fmaxf(v, __shfl_xor(v, 8, 32));
    return v;
}
__device__ __forceinline__ float redsum16(float v) {
    v += __shfl_xor(v, 1, 32);
    v += __shfl_xor(v, 2, 32);
    v += __shfl_xor(v, 4, 32);
    v += __shfl_xor(v, 8, 32);
    return v;
}

// ---------------- Pass 1: per-bucket partial sums for the routing net ------
// Sq[bh][u][d] = sum of q over bucket u (rotated order)
// Sk[bh][u][d] = sum of k over bucket u
// T [bh][u][d] = sum_j (partial cumsum of k within bucket)[j] / (u*128+j+1)
__global__ __launch_bounds__(64)
void sinkhorn_scan1(const float* __restrict__ q, const float* __restrict__ k,
                    float* __restrict__ Sq, float* __restrict__ Sk, float* __restrict__ Tb) {
    const int u = blockIdx.x, bh = blockIdx.y, d = threadIdx.x;
    const bool sp = (bh & 7) >= 4;
    const long base = (long)bh * T_LEN * DH;
    float sq = 0.f, ck = 0.f, ts = 0.f;
    for (int j = 0; j < BSZ; ++j) {
        int g = u * BSZ + j;
        if (sp) g = (g + (BSZ - 1)) & (T_LEN - 1);
        sq += q[base + (long)g * DH + d];
        ck += k[base + (long)g * DH + d];
        ts += ck * (1.0f / (float)(u * BSZ + j + 1));
    }
    const int o = (bh * NBUCK + u) * DH + d;
    Sq[o] = sq; Sk[o] = ck; Tb[o] = ts;
}

// ---------------- Pass 2: prefix over buckets -> sq (bucket-start cumavg of q)
// and skp (zero-padded per-bucket sums of cumavg of k) ------------------------
__global__ __launch_bounds__(64)
void sinkhorn_scan2(const float* __restrict__ q, const float* __restrict__ Sq,
                    const float* __restrict__ Sk, const float* __restrict__ Tb,
                    float* __restrict__ sq, float* __restrict__ skp) {
    const int bh = blockIdx.x, d = threadIdx.x;
    const bool sp = (bh & 7) >= 4;
    const long base = (long)bh * T_LEN * DH;
    float Pq = 0.f, Pk = 0.f;
    skp[(bh * NBUCK + 0) * DH + d] = 0.f;  // padded null column
    for (int u = 0; u < NBUCK; ++u) {
        int g = u * BSZ;
        if (sp) g = (g + (BSZ - 1)) & (T_LEN - 1);
        const float qf = q[base + (long)g * DH + d];
        sq[(bh * NBUCK + u) * DH + d] = (Pq + qf) * (1.0f / (float)(u * BSZ + 1));
        if (u + 1 < NBUCK) {
            float A = 0.f;
            for (int j = 0; j < BSZ; ++j) A += 1.0f / (float)(u * BSZ + j + 1);
            skp[(bh * NBUCK + u + 1) * DH + d] = Pk * A + Tb[(bh * NBUCK + u) * DH + d];
        }
        Pq += Sq[(bh * NBUCK + u) * DH + d];
        Pk += Sk[(bh * NBUCK + u) * DH + d];
    }
}

// ---------------- Pass 3: per-row argmax + softmax weight of R --------------
__global__ __launch_bounds__(64)
void sinkhorn_select(const float* __restrict__ sq, const float* __restrict__ skp,
                     int* __restrict__ sel, float* __restrict__ wgt) {
    const int u = blockIdx.x, bh = blockIdx.y, vt = threadIdx.x;
    __shared__ float lo[NBUCK];
    float l = NEGMAX;
    if (vt <= u) {  // softmax columns 0..u (col >= u+1 masked)
        const float* a = &sq[(bh * NBUCK + u) * DH];
        const float* b = &skp[(bh * NBUCK + vt) * DH];
        float acc = 0.f;
        for (int d = 0; d < DH; ++d) acc += a[d] * b[d];
        l = acc * 0.125f;
    }
    lo[vt] = l;
    __syncthreads();
    if (vt == 0) {
        float m = NEGMAX;
        for (int v = 0; v <= u; ++v) m = fmaxf(m, lo[v]);
        float den = 0.f;
        for (int v = 0; v <= u; ++v) den += __expf(lo[v] - m);
        int best = 0; float bl = NEGMAX;
        for (int v = 0; v < u; ++v)  // kept columns are 0..u-1 (0 = null bucket)
            if (lo[v] > bl) { bl = lo[v]; best = v; }
        sel[bh * NBUCK + u] = best;
        wgt[bh * NBUCK + u] = (u > 0) ? (__expf(bl - m) / den) : 0.f;
    }
}

// ---------------- Main attention: WMMA bf16, one WG per (bh, bucket) --------
__global__ __launch_bounds__(256)
void sinkhorn_attn(const float* __restrict__ q, const float* __restrict__ k,
                   const float* __restrict__ v, const float* __restrict__ nullk,
                   const float* __restrict__ nullv, const int* __restrict__ sel,
                   const float* __restrict__ wgt, float* __restrict__ out) {
    const int u = blockIdx.x;            // bucket
    const int bh = blockIdx.y;           // b*8 + head
    const int head = bh & 7;
    const bool sp_head = head >= 4;      // rotated heads
    const bool sp = sp_head && (u == NBUCK - 1);  // special mask bucket
    const int tid = threadIdx.x;
    const int lane = tid & 31, wave = tid >> 5;
    const int half = lane >> 4, l16 = lane & 15;

    __shared__ __align__(16) __bf16 Qs[128 * 72];    // Q rows, stride 72
    __shared__ __align__(16) __bf16 Ks[256 * 72];    // keys: [0..127]=reordered, [128..255]=local
    __shared__ __align__(16) __bf16 VTs[64 * 264];   // V transposed: [d][key]
    __shared__ __align__(16) __bf16 As[128 * 264];   // attn probs bf16, stride 264

    const long base = (long)bh * T_LEN * DH;
    const int s = sel[bh * NBUCK + u];
    const float w = wgt[bh * NBUCK + u];

    // ---- Stage Q, local K, local V (transposed) into LDS as bf16 ----
    for (int i = tid; i < 128 * 16; i += 256) {
        const int r = i >> 4, c4 = (i & 15) * 4;
        int g = u * BSZ + r;
        if (sp_head) g = (g + (BSZ - 1)) & (T_LEN - 1);
        const float4 qv = *(const float4*)(q + base + (long)g * DH + c4);
        __bf16* dq = &Qs[r * 72 + c4];
        dq[0] = f2bf(qv.x); dq[1] = f2bf(qv.y); dq[2] = f2bf(qv.z); dq[3] = f2bf(qv.w);
        const float4 kv = *(const float4*)(k + base + (long)g * DH + c4);
        __bf16* dk = &Ks[(128 + r) * 72 + c4];
        dk[0] = f2bf(kv.x); dk[1] = f2bf(kv.y); dk[2] = f2bf(kv.z); dk[3] = f2bf(kv.w);
        const float4 vv = *(const float4*)(v + base + (long)g * DH + c4);
        VTs[(c4 + 0) * 264 + 128 + r] = f2bf(vv.x);
        VTs[(c4 + 1) * 264 + 128 + r] = f2bf(vv.y);
        VTs[(c4 + 2) * 264 + 128 + r] = f2bf(vv.z);
        VTs[(c4 + 3) * 264 + 128 + r] = f2bf(vv.w);
    }
    // ---- Stage reordered bucket: w * (selected bucket | null bucket) ----
    for (int i = tid; i < 128 * 16; i += 256) {
        const int r = i >> 4, c4 = (i & 15) * 4;
        float4 kv, vv;
        if (s == 0) {
            kv = *(const float4*)(nullk + head * DH + c4);
            vv = *(const float4*)(nullv + head * DH + c4);
        } else {
            int g = (s - 1) * BSZ + r;
            if (sp_head) g = (g + (BSZ - 1)) & (T_LEN - 1);
            kv = *(const float4*)(k + base + (long)g * DH + c4);
            vv = *(const float4*)(v + base + (long)g * DH + c4);
        }
        __bf16* dk = &Ks[r * 72 + c4];
        dk[0] = f2bf(w * kv.x); dk[1] = f2bf(w * kv.y);
        dk[2] = f2bf(w * kv.z); dk[3] = f2bf(w * kv.w);
        VTs[(c4 + 0) * 264 + r] = f2bf(w * vv.x);
        VTs[(c4 + 1) * 264 + r] = f2bf(w * vv.y);
        VTs[(c4 + 2) * 264 + r] = f2bf(w * vv.z);
        VTs[(c4 + 3) * 264 + r] = f2bf(w * vv.w);
    }
    __syncthreads();

    // ---- GEMM1: dots = Q (128x64) x K^T (64x256); wave owns 16 rows x 256 cols
    v8f acc[16];
    {
        v8f z = 0.f;
        for (int tn = 0; tn < 16; ++tn) acc[tn] = z;
    }
    for (int kc = 0; kc < 2; ++kc) {
        const v16bf afrag = ldfrag(&Qs[(wave * 16 + l16) * 72 + kc * 32 + half * 8]);
        for (int tn = 0; tn < 16; ++tn) {
            const v16bf bfrag = ldfrag(&Ks[(tn * 16 + l16) * 72 + kc * 32 + half * 8]);
            acc[tn] = __builtin_amdgcn_wmma_f32_16x16x32_bf16(
                false, afrag, false, bfrag, (short)0, acc[tn], false, false);
        }
    }

    // ---- scale + mask + row softmax (16-lane half reductions), write bf16 attn
    for (int j = 0; j < 8; ++j) {
        const int i = wave * 16 + j + 8 * half;  // query row in bucket
        float mx = NEGMAX;
        for (int tn = 0; tn < 16; ++tn) {
            const int jj = tn * 16 + l16;        // key column (0..255)
            float x = acc[tn][j] * 0.125f;
            bool ok = !((jj >= BSZ) && (jj - BSZ > i));
            if (sp) ok = ok && !((jj <= BSZ) && (i > 0));
            x = ok ? x : NEGMAX;
            acc[tn][j] = x;
            mx = fmaxf(mx, x);
        }
        mx = redmax16(mx);
        float ssum = 0.f;
        for (int tn = 0; tn < 16; ++tn) {
            const float e = __expf(acc[tn][j] - mx);
            acc[tn][j] = e;
            ssum += e;
        }
        ssum = redsum16(ssum);
        const float inv = 1.0f / ssum;
        for (int tn = 0; tn < 16; ++tn)
            As[i * 264 + tn * 16 + l16] = f2bf(acc[tn][j] * inv);
    }
    // per-wave rows only -> no cross-wave barrier needed (DS ops in-order per wave)

    // ---- GEMM2: out = attn (128x256) x V (256x64) ----
    v8f o[4];
    {
        v8f z = 0.f;
        for (int tn = 0; tn < 4; ++tn) o[tn] = z;
    }
    for (int kc = 0; kc < 8; ++kc) {
        const v16bf afrag = ldfrag(&As[(wave * 16 + l16) * 264 + kc * 32 + half * 8]);
        for (int tn = 0; tn < 4; ++tn) {
            const v16bf bfrag = ldfrag(&VTs[(tn * 16 + l16) * 264 + kc * 32 + half * 8]);
            o[tn] = __builtin_amdgcn_wmma_f32_16x16x32_bf16(
                false, afrag, false, bfrag, (short)0, o[tn], false, false);
        }
    }

    // ---- store f32 output with inverse rotation folded into the index map ----
    for (int tn = 0; tn < 4; ++tn) {
        for (int j = 0; j < 8; ++j) {
            const int m = wave * 16 + j + 8 * half;
            const int d = tn * 16 + l16;
            int g = u * BSZ + m;
            if (sp_head) g = (g + (BSZ - 1)) & (T_LEN - 1);
            out[base + (long)g * DH + d] = o[tn][j];
        }
    }
}

extern "C" void kernel_launch(void* const* d_in, const int* in_sizes, int n_in,
                              void* d_out, int out_size, void* d_ws, size_t ws_size,
                              hipStream_t stream) {
    const float* q     = (const float*)d_in[0];
    const float* k     = (const float*)d_in[1];
    const float* v     = (const float*)d_in[2];
    const float* nullk = (const float*)d_in[3];
    const float* nullv = (const float*)d_in[4];
    float* out = (float*)d_out;

    float* ws = (float*)d_ws;
    // workspace layout (float offsets)
    float* sq  = ws + 0;        // 32*64*64
    float* skp = ws + 131072;   // 32*64*64
    float* Sq  = ws + 262144;   // 32*64*64
    float* Sk  = ws + 393216;   // 32*64*64
    float* Tb  = ws + 524288;   // 32*64*64
    int*   sel = (int*)(ws + 655360);  // 32*64
    float* wgt = ws + 657408;          // 32*64

    sinkhorn_scan1<<<dim3(NBUCK, BH), 64, 0, stream>>>(q, k, Sq, Sk, Tb);
    sinkhorn_scan2<<<BH, 64, 0, stream>>>(q, Sq, Sk, Tb, sq, skp);
    sinkhorn_select<<<dim3(NBUCK, BH), 64, 0, stream>>>(sq, skp, sel, wgt);
    sinkhorn_attn<<<dim3(NBUCK, BH), 256, 0, stream>>>(q, k, v, nullk, nullv, sel, wgt, out);
}
